// LocalMQA_80307298500855
// MI455X (gfx1250) — matrix operand order repaired and checked by reading, not compiled, not run on hardware
//
#include <hip/hip_runtime.h>

typedef _Float16 f16;
typedef _Float16 v4h  __attribute__((ext_vector_type(4)));
typedef _Float16 v8h  __attribute__((ext_vector_type(8)));
typedef _Float16 v16h __attribute__((ext_vector_type(16)));
typedef float    v8f  __attribute__((ext_vector_type(8)));
typedef int      v4i  __attribute__((ext_vector_type(4)));

#define SEQ    2048
#define DMODEL 1024
#define NH     16
#define HD     64
#define WIN    128
#define NQKV   1152   // HD*(NH+2)

// ---------------------------------------------------------------------------
// CDNA5 async copy helpers (ASYNCcnt-tracked global -> LDS DMA)
// ---------------------------------------------------------------------------
__device__ __forceinline__ void async_ld16(const f16* g, f16* l) {
#if __has_builtin(__builtin_amdgcn_global_load_async_to_lds_b128)
  __builtin_amdgcn_global_load_async_to_lds_b128(
      (v4i*)const_cast<f16*>(g), (v4i*)l, 0, 0);
#else
  unsigned loff = (unsigned)(size_t)(__attribute__((address_space(3))) f16*)l;
  asm volatile("global_load_async_to_lds_b128 %0, %1, off"
               :: "v"(loff), "v"(g) : "memory");
#endif
}

template <int N>
__device__ __forceinline__ void wait_async() {
#if __has_builtin(__builtin_amdgcn_s_wait_asynccnt)
  __builtin_amdgcn_s_wait_asynccnt(N);
#else
  asm volatile("s_wait_asynccnt %0" :: "i"(N) : "memory");
#endif
}

// ---------------------------------------------------------------------------
// Fragment helpers. Layouts per CDNA5 ISA 7.12.2 (wave32, 16-bit operands):
//  A 16x32 (MxK): lanes 0-15 -> M=lane, K 0..7 & 16..23; lanes 16-31 -> K 8..15 & 24..31
//  B 32x16 (KxN): lanes 0-15 -> N=lane, K 0..7 & 16..23; lanes 16-31 -> K 8..15 & 24..31
// ---------------------------------------------------------------------------
__device__ __forceinline__ v16h cat8(v8h lo, v8h hi) {
  return __builtin_shufflevector(lo, hi, 0,1,2,3,4,5,6,7,8,9,10,11,12,13,14,15);
}

__device__ __forceinline__ v16h load_afrag(const f16* __restrict__ src, int ld, int lane) {
  int hh = lane >> 4;
  const f16* p = src + (size_t)(lane & 15) * ld + 8 * hh;
  v8h lo = *(const v8h*)(p);
  v8h hi = *(const v8h*)(p + 16);
  return cat8(lo, hi);
}

__device__ __forceinline__ v16h load_bfrag(const f16* __restrict__ src, int ld, int lane) {
  int hh = lane >> 4;
  const f16* p = src + (size_t)(lane & 15) * ld + 8 * hh;
  v8h lo = *(const v8h*)(p);
  v8h hi = *(const v8h*)(p + 16);
  return cat8(lo, hi);
}

// ---------------------------------------------------------------------------
// f32 -> f16 conversion (vec4)
// ---------------------------------------------------------------------------
__global__ void f32_to_f16_v4(const float* __restrict__ in, f16* __restrict__ out, int n4) {
  int idx = blockIdx.x * blockDim.x + threadIdx.x;
  if (idx < n4) {
    float4 v = ((const float4*)in)[idx];
    v4h o; o.x = (f16)v.x; o.y = (f16)v.y; o.z = (f16)v.z; o.w = (f16)v.w;
    ((v4h*)out)[idx] = o;
  }
}

// vT[d][j] = v[j][d]  (v = qkv cols 64..127)
__global__ void build_vT(const f16* __restrict__ qkvh, f16* __restrict__ vT) {
  int idx = blockIdx.x * blockDim.x + threadIdx.x;  // over SEQ*HD
  int j = idx >> 6;
  int d = idx & 63;
  vT[(size_t)d * SEQ + j] = qkvh[(size_t)j * NQKV + HD + d];
}

// ---------------------------------------------------------------------------
// C = A(MxK) * B(NxK)^T + bias
// Block = 4 waves, 128x128 C tile; K-step 32 double-buffered through LDS via
// GLOBAL_LOAD_ASYNC_TO_LDS_B128 (ASYNCcnt). LDS rows padded 32->40 halves.
// ---------------------------------------------------------------------------
#define LDP 40  // padded row length in halves (80B -> conflict-free frag reads)

template<bool F32OUT>
__global__ __launch_bounds__(128)
void gemm_nt(const f16* __restrict__ A, const f16* __restrict__ B,
             const float* __restrict__ bias, void* __restrict__ Cout,
             int M, int N, int K) {
  __shared__ __align__(16) f16 smem[2][2][128 * LDP];  // [buf][A/B][row*LDP+col]

  int lane = threadIdx.x & 31;
  int wid  = threadIdx.x >> 5;
  int mblk = blockIdx.y * 128;
  int nblk = blockIdx.x * 128;
  int m0w  = (wid >> 1) * 64;          // wave's M offset within block
  int n0w  = (wid & 1) * 64;           // wave's N offset within block

  // cooperative async copy of one 128x32 A tile + 128x32 B tile
  auto issue_tile = [&](int k0, int buf) {
    int rbase = 32 * wid + (lane >> 2);   // wave covers 32 rows
    int cc = (lane & 3) * 8;              // 8-half chunk within k-step
#pragma unroll
    for (int q = 0; q < 4; ++q) {
      int row = rbase + 8 * q;
      async_ld16(A + (size_t)(mblk + row) * K + k0 + cc,
                 &smem[buf][0][row * LDP + cc]);
      async_ld16(B + (size_t)(nblk + row) * K + k0 + cc,
                 &smem[buf][1][row * LDP + cc]);
    }
  };

  v8f acc[4][4] = {};
  int nl = lane & 15, hh = lane >> 4;

  issue_tile(0, 0);
  int cur = 0;
  for (int k0 = 0; k0 < K; k0 += 32) {
    bool more = (k0 + 32 < K);
    if (more) { issue_tile(k0 + 32, cur ^ 1); wait_async<8>(); }
    else      { wait_async<0>(); }
    __syncthreads();   // all waves' DMAs for buf `cur` have landed

    v16h af[4], bf[4];
#pragma unroll
    for (int mi = 0; mi < 4; ++mi) {
      const f16* p = &smem[cur][0][(m0w + 16 * mi + nl) * LDP + 8 * hh];
      af[mi] = cat8(*(const v8h*)p, *(const v8h*)(p + 16));
    }
#pragma unroll
    for (int ni = 0; ni < 4; ++ni) {
      const f16* p = &smem[cur][1][(n0w + 16 * ni + nl) * LDP + 8 * hh];
      bf[ni] = cat8(*(const v8h*)p, *(const v8h*)(p + 16));
    }
#pragma unroll
    for (int mi = 0; mi < 4; ++mi)
#pragma unroll
      for (int ni = 0; ni < 4; ++ni)
        acc[mi][ni] = __builtin_amdgcn_wmma_f32_16x16x32_f16(
            false, af[mi], false, bf[ni], (short)0, acc[mi][ni], false, false);

    __syncthreads();   // protect buf `cur^1` from next iteration's DMA
    cur ^= 1;
  }

#pragma unroll
  for (int ni = 0; ni < 4; ++ni) {
    int n = nblk + n0w + 16 * ni + nl;
    float bv = bias[n];
#pragma unroll
    for (int mi = 0; mi < 4; ++mi) {
#pragma unroll
      for (int v = 0; v < 8; ++v) {
        int m = mblk + m0w + 16 * mi + v + 8 * hh;
        float val = acc[mi][ni][v] + bv;
        if (F32OUT) ((float*)Cout)[(size_t)m * N + n] = val;
        else        ((f16*)Cout)[(size_t)m * N + n] = (f16)val;
      }
    }
  }
}

// ---------------------------------------------------------------------------
// Sliding-window MQA, flash-style. One wave per (16-query block, head).
// scores = 8 * q.k ; window (i-128, i].
// ---------------------------------------------------------------------------
__global__ __launch_bounds__(32)
void attn_kernel(const f16* __restrict__ qkvh, const f16* __restrict__ vT,
                 f16* __restrict__ oh) {
  __shared__ __align__(16) f16 Pbuf[16 * 32];
  int lane = threadIdx.x & 31;
  int nl = lane & 15, hh = lane >> 4;
  int i0 = blockIdx.x * 16;
  int h  = blockIdx.y;

  const f16* qbase = qkvh + (size_t)i0 * NQKV + 2 * HD + h * HD;
  v16h aq0 = load_afrag(qbase + 0,  NQKV, lane);
  v16h aq1 = load_afrag(qbase + 32, NQKV, lane);

  float mrow[8], lrow[8];
  v8f oacc[4] = {};
#pragma unroll
  for (int v = 0; v < 8; ++v) { mrow[v] = -1e4f; lrow[v] = 0.0f; }

  int js = (i0 >= WIN) ? ((i0 - WIN + 1) & ~31) : 0;
  for (int j0 = js; j0 <= i0 + 15; j0 += 32) {
    v8f s[2] = {};
#pragma unroll
    for (int t = 0; t < 2; ++t) {
      const f16* kb = qkvh + (size_t)(j0 + 16 * t) * NQKV;
      v16h b0 = load_bfrag(kb + 0,  NQKV, lane);
      v16h b1 = load_bfrag(kb + 32, NQKV, lane);
      s[t] = __builtin_amdgcn_wmma_f32_16x16x32_f16(false, aq0, false, b0, (short)0, s[t], false, false);
      s[t] = __builtin_amdgcn_wmma_f32_16x16x32_f16(false, aq1, false, b1, (short)0, s[t], false, false);
    }

    float rmax[8];
#pragma unroll
    for (int v = 0; v < 8; ++v) rmax[v] = -1e4f;
#pragma unroll
    for (int t = 0; t < 2; ++t) {
      int j = j0 + 16 * t + nl;
#pragma unroll
      for (int v = 0; v < 8; ++v) {
        int i = i0 + v + 8 * hh;
        bool ok = (j <= i) && (j > i - WIN);
        float val = ok ? 8.0f * s[t][v] : -1e4f;
        s[t][v] = val;
        rmax[v] = fmaxf(rmax[v], val);
      }
    }
#pragma unroll
    for (int off = 8; off >= 1; off >>= 1)
#pragma unroll
      for (int v = 0; v < 8; ++v)
        rmax[v] = fmaxf(rmax[v], __shfl_xor(rmax[v], off, 32));

    float alpha[8], rsum[8];
#pragma unroll
    for (int v = 0; v < 8; ++v) {
      float mn = fmaxf(mrow[v], rmax[v]);
      alpha[v] = __expf(mrow[v] - mn);
      mrow[v] = mn;
      rsum[v] = 0.0f;
    }

#pragma unroll
    for (int t = 0; t < 2; ++t) {
#pragma unroll
      for (int v = 0; v < 8; ++v) {
        float p = __expf(s[t][v] - mrow[v]);
        rsum[v] += p;
        Pbuf[(v + 8 * hh) * 32 + 16 * t + nl] = (f16)p;
      }
    }
#pragma unroll
    for (int off = 8; off >= 1; off >>= 1)
#pragma unroll
      for (int v = 0; v < 8; ++v)
        rsum[v] += __shfl_xor(rsum[v], off, 32);
#pragma unroll
    for (int v = 0; v < 8; ++v)
      lrow[v] = lrow[v] * alpha[v] + rsum[v];

    __syncthreads();
    v16h pf;
    {
      const f16* p = Pbuf + nl * 32 + 8 * hh;
      v8h lo = *(const v8h*)(p);
      v8h hi = *(const v8h*)(p + 16);
      pf = cat8(lo, hi);
    }

#pragma unroll
    for (int dt = 0; dt < 4; ++dt) {
      v16h bv = load_bfrag(vT + (size_t)(dt * 16) * SEQ + j0, SEQ, lane);
#pragma unroll
      for (int v = 0; v < 8; ++v) oacc[dt][v] *= alpha[v];
      oacc[dt] = __builtin_amdgcn_wmma_f32_16x16x32_f16(
          false, pf, false, bv, (short)0, oacc[dt], false, false);
    }
    __syncthreads();
  }

#pragma unroll
  for (int dt = 0; dt < 4; ++dt) {
    int d = h * HD + dt * 16 + nl;
#pragma unroll
    for (int v = 0; v < 8; ++v) {
      int i = i0 + v + 8 * hh;
      oh[(size_t)i * DMODEL + d] = (f16)(oacc[dt][v] / lrow[v]);
    }
  }
}

// ---------------------------------------------------------------------------
extern "C" void kernel_launch(void* const* d_in, const int* in_sizes, int n_in,
                              void* d_out, int out_size, void* d_ws, size_t ws_size,
                              hipStream_t stream) {
  const float* x    = (const float*)d_in[0];
  const float* Wqkv = (const float*)d_in[1];
  const float* bqkv = (const float*)d_in[2];
  const float* Wout = (const float*)d_in[3];
  const float* bout = (const float*)d_in[4];

  char* ws = (char*)d_ws;
  f16* xh    = (f16*)ws; ws += (size_t)SEQ * DMODEL * 2;
  f16* wqkvh = (f16*)ws; ws += (size_t)NQKV * DMODEL * 2;
  f16* wouth = (f16*)ws; ws += (size_t)DMODEL * DMODEL * 2;
  f16* qkvh  = (f16*)ws; ws += (size_t)SEQ * NQKV * 2;
  f16* vT    = (f16*)ws; ws += (size_t)HD * SEQ * 2;
  f16* oh    = (f16*)ws; ws += (size_t)SEQ * DMODEL * 2;

  {
    int n4 = SEQ * DMODEL / 4;
    f32_to_f16_v4<<<(n4 + 255) / 256, 256, 0, stream>>>(x, xh, n4);
    n4 = NQKV * DMODEL / 4;
    f32_to_f16_v4<<<(n4 + 255) / 256, 256, 0, stream>>>(Wqkv, wqkvh, n4);
    n4 = DMODEL * DMODEL / 4;
    f32_to_f16_v4<<<(n4 + 255) / 256, 256, 0, stream>>>(Wout, wouth, n4);
  }

  gemm_nt<false><<<dim3(NQKV / 128, SEQ / 128), 128, 0, stream>>>(
      xh, wqkvh, bqkv, (void*)qkvh, SEQ, NQKV, DMODEL);

  build_vT<<<(SEQ * HD) / 256, 256, 0, stream>>>(qkvh, vT);

  attn_kernel<<<dim3(SEQ / 16, NH), 32, 0, stream>>>(qkvh, vT, oh);

  gemm_nt<true><<<dim3(DMODEL / 128, SEQ / 128), 128, 0, stream>>>(
      oh, wouth, bout, d_out, SEQ, DMODEL, DMODEL);
}
